// GrowthLayer_34694745817206
// MI455X (gfx1250) — matrix operand order repaired and checked by reading, not compiled, not run on hardware
//
#include <hip/hip_runtime.h>
#include <hip/hip_bf16.h>

typedef __bf16 bf16_t;
typedef __attribute__((ext_vector_type(16))) __bf16 v16bf;
typedef __attribute__((ext_vector_type(8)))  __bf16 v8bf;
typedef __attribute__((ext_vector_type(4)))  __bf16 v4bf;
typedef __attribute__((ext_vector_type(8)))  float  v8f;
typedef __attribute__((ext_vector_type(4)))  unsigned int u32x4;
typedef __attribute__((ext_vector_type(8)))  int i32x8;
typedef __attribute__((ext_vector_type(4)))  int i32x4;

#define BATCH   8
#define TSEQ    4096
#define DMODEL  1024
#define NHEAD   16

#ifndef HAVE_TDM
#if defined(__has_builtin)
#if __has_builtin(__builtin_amdgcn_tensor_load_to_lds)
#define HAVE_TDM 1
#endif
#endif
#endif
#ifndef HAVE_TDM
#define HAVE_TDM 0
#endif

// ---------------------------------------------------------------------------
// Elementwise f32 -> bf16 convert (vectorized: 2x float4 in, 1x v8bf out)
// ---------------------------------------------------------------------------
__global__ __launch_bounds__(256)
void cvt_f32_to_bf16_kernel(const float* __restrict__ in,
                            bf16_t* __restrict__ out, long long n) {
    long long i = ((long long)blockIdx.x * 256 + threadIdx.x) * 8;
    if (i + 8 <= n) {
        const float4 a = *(const float4*)(in + i);
        const float4 b = *(const float4*)(in + i + 4);
        v8bf o;
        o[0] = (bf16_t)a.x; o[1] = (bf16_t)a.y; o[2] = (bf16_t)a.z; o[3] = (bf16_t)a.w;
        o[4] = (bf16_t)b.x; o[5] = (bf16_t)b.y; o[6] = (bf16_t)b.z; o[7] = (bf16_t)b.w;
        *(v8bf*)(out + i) = o;
    } else {
        for (; i < n; ++i) out[i] = (bf16_t)in[i];
    }
}

// ---------------------------------------------------------------------------
// Weight convert + transpose: Wt[n][k] = bf16(W[k][n]).  32x32 tiles via LDS.
// ---------------------------------------------------------------------------
__global__ __launch_bounds__(256)
void transpose_cvt_kernel(const float* __restrict__ W, bf16_t* __restrict__ Wt) {
    __shared__ float tile[32][33];
    const int bk = blockIdx.x * 32;
    const int bn = blockIdx.y * 32;
    const int tx = threadIdx.x;               // 0..31
    const int ty = threadIdx.y;               // 0..7
    for (int i = 0; i < 32; i += 8)
        tile[ty + i][tx] = W[(size_t)(bk + ty + i) * DMODEL + bn + tx];
    __syncthreads();
    for (int i = 0; i < 32; i += 8)
        Wt[(size_t)(bn + ty + i) * DMODEL + bk + tx] = (bf16_t)tile[tx][ty + i];
}

// ---------------------------------------------------------------------------
// TDM 2D tile load descriptor (CDNA5 ISA):
//   group0: [1:0]count=1 | [63:32]lds_addr | [120:64]global_addr | [127:126]type=2
//   group1: data_size=2B, pad_enable (4 DWORDs pad per 32 DWORDs -> LDS row
//           stride 72 bf16 = 144B), tensor dims (remaining extent, OOB->0),
//           tile dims, dim0 stride (elements).
// ---------------------------------------------------------------------------
#if HAVE_TDM
__device__ __forceinline__ void tdm_load_2d(unsigned lds_addr, const void* gptr,
                                            unsigned rem_d0, unsigned rem_d1,
                                            unsigned tile_d0, unsigned tile_d1,
                                            unsigned stride0_elems) {
    const unsigned long long ga = (unsigned long long)(uintptr_t)gptr;
    u32x4 g0;
    g0[0] = 1u;                                              // count=1, user mode
    g0[1] = lds_addr;                                        // LDS byte address
    g0[2] = (unsigned)(ga & 0xffffffffull);                  // global_addr[31:0]
    g0[3] = (unsigned)((ga >> 32) & 0x01ffffffull) | (2u << 30); // addr[56:32] | type=2

    const unsigned dw0 = (1u << 16)    // data_size = 2 bytes
                       | (1u << 20)    // pad_enable
                       | (4u << 22)    // pad_interval: 32 DWORDs (code 4)
                       | (3u << 25);   // pad_amount:   4 DWORDs (code 3)
    const unsigned dw1 = (rem_d0 & 0xffffu) << 16;               // tensor_dim0 lo
    const unsigned dw2 = (rem_d0 >> 16) | ((rem_d1 & 0xffffu) << 16);
    const unsigned dw3 = (rem_d1 >> 16) | (tile_d0 << 16);       // tile_dim0
    const unsigned dw4 = tile_d1;                                // tile_dim1 (tile_dim2=0)
    const unsigned dw5 = stride0_elems;                          // tensor_dim0_stride lo32
    i32x8 g1 = {(int)dw0, (int)dw1, (int)dw2, (int)dw3,
                (int)dw4, (int)dw5, 0, 0};
    i32x4 z4 = {0, 0, 0, 0};
#if __clang_major__ >= 23
    i32x8 z8 = {0, 0, 0, 0, 0, 0, 0, 0};
    __builtin_amdgcn_tensor_load_to_lds(g0, g1, z4, z4, z8, 0);
#else
    __builtin_amdgcn_tensor_load_to_lds(g0, g1, z4, z4, 0);
#endif
}
#endif

// ---------------------------------------------------------------------------
// Tiled WMMA GEMM:  C[M,N] = A[M,K] * Bt[N,K]^T + bias[N]    (A, Bt bf16)
//   Block tile 128x128, BK=64 (two WMMA K-steps per stage), 8 waves (2x4),
//   wave tile 64x32 -> 16 WMMA per wave per stage.
//   Double-buffered LDS; tiles staged by the Tensor Data Mover (wave 0 issues,
//   TENSORcnt + one workgroup barrier per stage; next-stage TDM overlaps
//   compute).  Falls back to manual global->LDS staging without TDM support.
// ---------------------------------------------------------------------------
__global__ __launch_bounds__(256)
void wmma_gemm_kernel(const bf16_t* __restrict__ A,
                      const bf16_t* __restrict__ Bt,
                      const float* __restrict__ bias,
                      float* __restrict__ C,
                      int M, int N, int K) {
    constexpr int BM = 128, BN = 128, BK = 64;
    constexpr int LDT = BK + 8;                 // 72 bf16 = 144B row stride (16B aligned)
    __shared__ bf16_t Alds[2][BM][LDT];
    __shared__ bf16_t Blds[2][BN][LDT];

    const int tid  = threadIdx.x;
    const int lane = tid & 31;
    const int wave = tid >> 5;                  // 0..7
    const int wm   = wave & 1;
    const int wn   = wave >> 1;
    const int half = lane >> 4;                 // lane half
    const int lrow = lane & 15;
    const int kfA  = half * 8;                  // A frag K base {0-7,16-23}/{8-15,24-31}
    const int kfB  = half * 8;                  // B frag K base {0-7,16-23}/{8-15,24-31}

    const int m0 = blockIdx.y * BM;
    const int n0 = blockIdx.x * BN;
    const int nIter = K / BK;

    const v8f vzero = {0.f, 0.f, 0.f, 0.f, 0.f, 0.f, 0.f, 0.f};
    v8f acc[4][2];
    for (int i = 0; i < 4; ++i)
        for (int j = 0; j < 2; ++j)
            acc[i][j] = vzero;

#if HAVE_TDM
    auto issue_stage = [&](int it, int buf) {
        const int k0 = it * BK;
        tdm_load_2d((unsigned)(uintptr_t)&Alds[buf][0][0],
                    A + (size_t)m0 * K + k0,
                    (unsigned)(K - k0), (unsigned)(M - m0), BK, BM, (unsigned)K);
        tdm_load_2d((unsigned)(uintptr_t)&Blds[buf][0][0],
                    Bt + (size_t)n0 * K + k0,
                    (unsigned)(K - k0), (unsigned)(N - n0), BK, BN, (unsigned)K);
    };
    if (tid < 32) issue_stage(0, 0);
#endif

    for (int it = 0; it < nIter; ++it) {
        const int cur = it & 1;
#if HAVE_TDM
        if (tid < 32) __builtin_amdgcn_s_wait_tensorcnt(0);
        __syncthreads();                          // buf[cur] visible; buf[cur^1] free
        if (tid < 32 && it + 1 < nIter) issue_stage(it + 1, cur ^ 1);
#else
        const int k0 = it * BK;
        __syncthreads();                          // prev compute done before overwrite
        #pragma unroll
        for (int i = 0; i < 4; ++i) {             // A tile: 128 x 64 bf16
            const int idx = tid + i * 256;        // 0..1023
            const int r   = idx >> 3;
            const int c8  = (idx & 7) * 8;
            const int gm  = m0 + r;
            v8bf v;
            #pragma unroll
            for (int q = 0; q < 8; ++q) v[q] = (bf16_t)0.0f;
            if (gm < M) v = *(const v8bf*)&A[(size_t)gm * K + k0 + c8];
            *(v8bf*)&Alds[cur][r][c8] = v;
        }
        #pragma unroll
        for (int i = 0; i < 4; ++i) {             // B tile: 128 x 64 bf16
            const int idx = tid + i * 256;
            const int r   = idx >> 3;
            const int c8  = (idx & 7) * 8;
            v8bf v = *(const v8bf*)&Bt[(size_t)(n0 + r) * K + k0 + c8];
            *(v8bf*)&Blds[cur][r][c8] = v;
        }
        __syncthreads();
#endif
        // ---- compute: 2 WMMA K-steps, 8 WMMA each, per wave ----
        #pragma unroll
        for (int s = 0; s < 2; ++s) {
            const int ks = s * 32;
            v16bf afr[4], bfr[2];
            #pragma unroll
            for (int i = 0; i < 4; ++i) {
                const int r = wm * 64 + i * 16 + lrow;
                union { v16bf v; v8bf h[2]; } u;
                u.h[0] = *(const v8bf*)&Alds[cur][r][ks + kfA];
                u.h[1] = *(const v8bf*)&Alds[cur][r][ks + kfA + 16];
                afr[i] = u.v;
            }
            #pragma unroll
            for (int j = 0; j < 2; ++j) {
                const int c = wn * 32 + j * 16 + lrow;
                union { v16bf v; v8bf h[2]; } u;
                u.h[0] = *(const v8bf*)&Blds[cur][c][ks + kfB];
                u.h[1] = *(const v8bf*)&Blds[cur][c][ks + kfB + 16];
                bfr[j] = u.v;
            }
            #pragma unroll
            for (int i = 0; i < 4; ++i)
                #pragma unroll
                for (int j = 0; j < 2; ++j)
                    acc[i][j] = __builtin_amdgcn_wmma_f32_16x16x32_bf16(
                        false, afr[i], false, bfr[j], (short)0, acc[i][j],
                        false, false);
        }
    }

    __syncthreads();
    // ---- epilogue: bias add + f32 store ----
    #pragma unroll
    for (int i = 0; i < 4; ++i) {
        #pragma unroll
        for (int j = 0; j < 2; ++j) {
            const int gn = n0 + wn * 32 + j * 16 + lrow;
            const float bv = bias[gn];
            #pragma unroll
            for (int r = 0; r < 8; ++r) {
                const int gm = m0 + wm * 64 + i * 16 + half * 8 + r;
                if (gm < M) C[(size_t)gm * N + gn] = acc[i][j][r] + bv;
            }
        }
    }
}

// ---------------------------------------------------------------------------
// Fused temporal-diff + exponential smoothing scan; emits bf16 rows (with the
// prepended v0 row) directly as GEMM2's A matrix.  One block per batch, one
// thread per channel; coalesced row sweeps.
// ---------------------------------------------------------------------------
__global__ __launch_bounds__(1024)
void diff_scan_kernel(const float* __restrict__ V,
                      const float* __restrict__ z0,
                      const float* __restrict__ v0,
                      const float* __restrict__ slogit,
                      bf16_t* __restrict__ S) {
    const int b  = blockIdx.x;
    const int ch = threadIdx.x;                 // 0..1023
    const int h  = ch >> 6;
    const float a  = 1.0f / (1.0f + __expf(-slogit[h]));
    const float om = 1.0f - a;

    float prev = z0[ch];
    float s    = v0[ch];

    const float* Vb = V + (size_t)b * TSEQ * DMODEL + ch;
    bf16_t*      Sb = S + (size_t)b * (TSEQ + 1) * DMODEL + ch;

    Sb[0] = (bf16_t)s;                          // prepended v0 row
    for (int t = 0; t < TSEQ; ++t) {
        const float v  = Vb[(size_t)t * DMODEL];
        const float dv = v - prev;
        prev = v;
        s = a * s + om * dv;
        Sb[(size_t)(t + 1) * DMODEL] = (bf16_t)s;
    }
}

// ---------------------------------------------------------------------------
extern "C" void kernel_launch(void* const* d_in, const int* in_sizes, int n_in,
                              void* d_out, int out_size, void* d_ws, size_t ws_size,
                              hipStream_t stream) {
    const float* inputs = (const float*)d_in[0];   // (B,T,1024)
    const float* z0     = (const float*)d_in[1];   // (16,64)
    const float* W_in   = (const float*)d_in[2];   // (1024,1024)
    const float* b_in   = (const float*)d_in[3];   // (1024,)
    const float* W_out  = (const float*)d_in[4];   // (1024,1024)
    const float* b_out  = (const float*)d_in[5];   // (1024,)
    const float* slog   = (const float*)d_in[6];   // (16,1)
    const float* v0     = (const float*)d_in[7];   // (1,1,16,64)
    float* out = (float*)d_out;                    // (B,T+1,1024)

    const long long nX = (long long)BATCH * TSEQ * DMODEL;

    // workspace layout
    char* ws = (char*)d_ws;
    bf16_t* Wt_in  = (bf16_t*)ws;                                    //   2 MB
    bf16_t* Wt_out = (bf16_t*)(ws + (2ull << 20));                   //   2 MB
    bf16_t* Xbf    = (bf16_t*)(ws + (4ull << 20));                   //  64 MB
    float*  V      = (float*)(ws + (4ull << 20) + 2ull * nX);        // 128 MB
    bf16_t* S      = (bf16_t*)(ws + (4ull << 20) + 6ull * nX);       //  64 MB

    const dim3 tb(32, 8);
    transpose_cvt_kernel<<<dim3(32, 32), tb, 0, stream>>>(W_in,  Wt_in);
    transpose_cvt_kernel<<<dim3(32, 32), tb, 0, stream>>>(W_out, Wt_out);
    cvt_f32_to_bf16_kernel<<<(int)(nX / 2048), 256, 0, stream>>>(inputs, Xbf, nX);

    {   // GEMM1: V = Xbf @ W_in + b_in   (M = 32768)
        const int M = BATCH * TSEQ;
        dim3 grid(DMODEL / 128, (M + 127) / 128);
        wmma_gemm_kernel<<<grid, 256, 0, stream>>>(Xbf, Wt_in, b_in, V,
                                                   M, DMODEL, DMODEL);
    }

    diff_scan_kernel<<<BATCH, DMODEL, 0, stream>>>(V, z0, v0, slog, S);

    {   // GEMM2: out = S @ W_out + b_out    (M = 32776, tail handled by TDM OOB)
        const int M = BATCH * (TSEQ + 1);
        dim3 grid(DMODEL / 128, (M + 127) / 128);
        wmma_gemm_kernel<<<grid, 256, 0, stream>>>(S, Wt_out, b_out, out,
                                                   M, DMODEL, DMODEL);
    }
}
